// HyenaBlock_89842125897748
// MI455X (gfx1250) — matrix operand order repaired and checked by reading, not compiled, not run on hardware
//
#include <hip/hip_runtime.h>
#include <hip/hip_bf16.h>

// ---------------------------------------------------------------------------
// Hyena block: u = xWu+bu ; v = xWv+bv ; v_conv = causal_conv(v, h_filter*decay)
//              y = (u * v_conv) Wo + bo
// GEMMs: bf16 WMMA (v_wmma_f32_16x16x32_bf16) with double-buffered
//        global_load_async_to_lds_b128 staging (ASYNCcnt pipeline).
// Conv:  f32 VALU, LDS-tiled time domain (causality via s<0 zero-guard).
// ---------------------------------------------------------------------------

typedef __attribute__((ext_vector_type(16))) __bf16 v16bf;
typedef __attribute__((ext_vector_type(8)))  __bf16 bf16x8;
typedef __attribute__((ext_vector_type(4)))  __bf16 bf16x4;
typedef __attribute__((ext_vector_type(8)))  float  v8f;

#define HB_B 4
#define HB_L 2048
#define HB_D 1024
#define HB_M (HB_B * HB_L)   // 8192 rows for the GEMMs

// ---------------------------------------------------------------------------
// Prep kernels
// ---------------------------------------------------------------------------

// f32 -> bf16, 4 elements per thread
__global__ void k_cvt_bf16(const float* __restrict__ src, __bf16* __restrict__ dst) {
    int i = (blockIdx.x * blockDim.x + threadIdx.x) * 4;
    float4 f = *(const float4*)(src + i);
    bf16x4 o;
    o.x = (__bf16)f.x; o.y = (__bf16)f.y; o.z = (__bf16)f.z; o.w = (__bf16)f.w;
    *(bf16x4*)(dst + i) = o;
}

// W[k][n] f32 -> WT[n][k] bf16, 32x32 LDS tiles (GEMM B-fragments become k-contiguous)
__global__ void k_transpose_cvt(const float* __restrict__ W, __bf16* __restrict__ WT) {
    __shared__ float tile[32][33];
    const int tx = threadIdx.x & 31;
    const int ty = threadIdx.x >> 5;           // 0..7
    const int nt = blockIdx.x, kt = blockIdx.y;
#pragma unroll
    for (int j = 0; j < 4; ++j) {
        int r = ty + j * 8;
        tile[r][tx] = W[(size_t)(kt * 32 + r) * HB_D + nt * 32 + tx];
    }
    __syncthreads();
#pragma unroll
    for (int j = 0; j < 4; ++j) {
        int r = ty + j * 8;
        WT[(size_t)(nt * 32 + r) * HB_D + kt * 32 + tx] = (__bf16)tile[tx][r];
    }
}

// h[t][d] = h_filter[t][d] * exp(-0.01 t)
__global__ void k_decay_h(const float* __restrict__ hf, float* __restrict__ h) {
    int i = blockIdx.x * blockDim.x + threadIdx.x;   // over L*D
    int t = i >> 10;                                  // D = 1024
    h[i] = hf[i] * __expf(-0.01f * (float)t);
}

// ---------------------------------------------------------------------------
// bf16 WMMA GEMM: C[M][N] = A[M][K] @ WT[N][K]^T + bias[N]
// Block: 256 threads (8 waves), 128x128 output tile, K-stage 64,
// double-buffered async global->LDS copies.
// ---------------------------------------------------------------------------
#define LDT2 72   // padded LDS row stride (bf16): 144B -> conflict-free b128 reads

// one wave issues 8 async b128 copies per (A+B) stage buffer
__device__ __forceinline__ void issue_tile_loads(const __bf16* __restrict__ A,
                                                 const __bf16* __restrict__ BT,
                                                 __bf16* As, __bf16* Bs,
                                                 int m0, int n0, int k0, int K,
                                                 int tid) {
#pragma unroll
    for (int i = 0; i < 4; ++i) {
        int idx = tid + i * 256;          // 0..1023
        int row = idx >> 3;               // 0..127
        int ck  = (idx & 7) * 8;          // 0,8,...,56 (bf16 elems)
        uint32_t aoff = (uint32_t)(uintptr_t)(&As[row * LDT2 + ck]);
        uint32_t boff = (uint32_t)(uintptr_t)(&Bs[row * LDT2 + ck]);
        uint64_t ag = (uint64_t)(uintptr_t)(A  + (size_t)(m0 + row) * K + k0 + ck);
        uint64_t bg = (uint64_t)(uintptr_t)(BT + (size_t)(n0 + row) * K + k0 + ck);
        asm volatile("global_load_async_to_lds_b128 %0, %1, off"
                     :: "v"(aoff), "v"(ag) : "memory");
        asm volatile("global_load_async_to_lds_b128 %0, %1, off"
                     :: "v"(boff), "v"(bg) : "memory");
    }
}

__global__ __launch_bounds__(256)
void k_gemm_bias(const __bf16* __restrict__ A, const __bf16* __restrict__ BT,
                 const float* __restrict__ bias, float* __restrict__ C,
                 int M, int N, int K) {
    __shared__ __bf16 Asm[2][128 * LDT2];
    __shared__ __bf16 Bsm[2][128 * LDT2];

    const int tid   = threadIdx.x;
    const int lane  = tid & 31;
    const int wid   = tid >> 5;          // 0..7
    const int waveM = wid & 3;           // 0..3  -> 32-row strip
    const int waveN = wid >> 2;          // 0..1  -> 64-col strip
    const int mrow  = lane & 15;
    const int half  = lane >> 4;         // 0 or 1

    const int m0 = blockIdx.y * 128;
    const int n0 = blockIdx.x * 128;

    v8f acc[2][4];
#pragma unroll
    for (int im = 0; im < 2; ++im)
#pragma unroll
        for (int in = 0; in < 4; ++in) acc[im][in] = (v8f)0.0f;

    union FragU { v16bf v; bf16x8 h[2]; };

    // prologue: stage 0 into buffer 0
    issue_tile_loads(A, BT, Asm[0], Bsm[0], m0, n0, 0, K, tid);

    int cur = 0;
    for (int k0 = 0; k0 < K; k0 += 64) {
        const bool has_next = (k0 + 64) < K;
        if (has_next)
            issue_tile_loads(A, BT, Asm[cur ^ 1], Bsm[cur ^ 1], m0, n0, k0 + 64, K, tid);

        // async copies complete in order: oldest 8 (current buffer) are done
        if (has_next) asm volatile("s_wait_asynccnt 0x8" ::: "memory");
        else          asm volatile("s_wait_asynccnt 0x0" ::: "memory");
        __syncthreads();

        const __bf16* As = Asm[cur];
        const __bf16* Bs = Bsm[cur];
#pragma unroll
        for (int ks = 0; ks < 64; ks += 32) {
            // fragments per ISA 16-bit A 16x32 / B 32x16 layouts
            v16bf afrag[2], bfrag[4];
#pragma unroll
            for (int im = 0; im < 2; ++im) {
                const __bf16* base = &As[(waveM * 32 + im * 16 + mrow) * LDT2 + ks];
                FragU u;
                u.h[0] = *(const bf16x8*)(base + half * 8);        // K 0..7  / 8..15
                u.h[1] = *(const bf16x8*)(base + half * 8 + 16);   // K 16..23/ 24..31
                afrag[im] = u.v;
            }
#pragma unroll
            for (int in = 0; in < 4; ++in) {
                const __bf16* base = &Bs[(waveN * 64 + in * 16 + mrow) * LDT2 + ks + half * 8];
                FragU u;
                u.h[0] = *(const bf16x8*)(base);
                u.h[1] = *(const bf16x8*)(base + 16);
                bfrag[in] = u.v;
            }
#pragma unroll
            for (int im = 0; im < 2; ++im)
#pragma unroll
                for (int in = 0; in < 4; ++in)
                    acc[im][in] = __builtin_amdgcn_wmma_f32_16x16x32_bf16(
                        false, afrag[im], false, bfrag[in],
                        (short)0, acc[im][in], false, false);
        }
        __syncthreads();   // all waves done reading before buffer is re-issued
        cur ^= 1;
    }

    // ---- epilogue: C/D layout m = r + 8*half, n = mrow ----
#pragma unroll
    for (int im = 0; im < 2; ++im) {
#pragma unroll
        for (int in = 0; in < 4; ++in) {
            int gm = m0 + waveM * 32 + im * 16 + half * 8;
            int gn = n0 + waveN * 64 + in * 16 + mrow;
            float bv = bias[gn];
#pragma unroll
            for (int r = 0; r < 8; ++r)
                C[(size_t)(gm + r) * N + gn] = acc[im][in][r] + bv;
        }
    }
}

// ---------------------------------------------------------------------------
// Causal conv + gating:  g[b,t,d] = bf16( u[b,t,d] * Σ_{τ≤t} h[τ,d] v[b,t-τ,d] )
// Block: 256 threads, tile = 64 t x 32 d, lag chunks of 64 through LDS.
// ---------------------------------------------------------------------------
__global__ __launch_bounds__(256)
void k_conv_gate(const float* __restrict__ v, const float* __restrict__ h,
                 const float* __restrict__ u, __bf16* __restrict__ g) {
    __shared__ float hs[64][33];
    __shared__ float vs[128][33];

    const int t0 = blockIdx.x * 64;
    const int d0 = blockIdx.y * 32;
    const int b  = blockIdx.z;
    const int dIdx = threadIdx.x & 31;
    const int tIdx = threadIdx.x >> 5;   // 0..7

    float acc[8];
#pragma unroll
    for (int j = 0; j < 8; ++j) acc[j] = 0.0f;

    for (int tau0 = 0; tau0 <= t0; tau0 += 64) {
        const int sbase = t0 - tau0 - 63;
#pragma unroll
        for (int j = 0; j < 8; ++j) {
            int row = tIdx + 8 * j;
            hs[row][dIdx] = h[(size_t)(tau0 + row) * HB_D + d0 + dIdx];
        }
#pragma unroll
        for (int j = 0; j < 16; ++j) {
            int row = tIdx + 8 * j;
            int s = sbase + row;
            vs[row][dIdx] = (s >= 0 && s < HB_L)
                                ? v[((size_t)b * HB_L + s) * HB_D + d0 + dIdx]
                                : 0.0f;
        }
        __syncthreads();

#pragma unroll 4
        for (int tt = 0; tt < 64; ++tt) {
            float hv = hs[tt][dIdx];
#pragma unroll
            for (int jj = 0; jj < 8; ++jj) {
                // t = t0 + tIdx + 8*jj ; srow = tIdx + 8*jj + 63 - tt
                acc[jj] += hv * vs[tIdx + 8 * jj + 63 - tt][dIdx];
            }
        }
        __syncthreads();
    }

#pragma unroll
    for (int jj = 0; jj < 8; ++jj) {
        int t = t0 + tIdx + 8 * jj;
        size_t idx = ((size_t)b * HB_L + t) * HB_D + d0 + dIdx;
        g[idx] = (__bf16)(u[idx] * acc[jj]);
    }
}

// ---------------------------------------------------------------------------
// Launch
// ---------------------------------------------------------------------------
extern "C" void kernel_launch(void* const* d_in, const int* in_sizes, int n_in,
                              void* d_out, int out_size, void* d_ws, size_t ws_size,
                              hipStream_t stream) {
    const float* x   = (const float*)d_in[0];
    const float* Wu  = (const float*)d_in[1];
    const float* bu  = (const float*)d_in[2];
    const float* Wv  = (const float*)d_in[3];
    const float* bv  = (const float*)d_in[4];
    const float* hf  = (const float*)d_in[5];
    const float* Wo  = (const float*)d_in[6];
    const float* bo  = (const float*)d_in[7];
    float* out = (float*)d_out;

    // workspace layout (bytes, 256-aligned)
    char* ws = (char*)d_ws;
    size_t off = 0;
    __bf16* xb  = (__bf16*)(ws + off); off += (size_t)HB_M * HB_D * 2;   // 16 MB
    __bf16* WuT = (__bf16*)(ws + off); off += (size_t)HB_D * HB_D * 2;   //  2 MB
    __bf16* WvT = (__bf16*)(ws + off); off += (size_t)HB_D * HB_D * 2;   //  2 MB
    __bf16* WoT = (__bf16*)(ws + off); off += (size_t)HB_D * HB_D * 2;   //  2 MB
    float*  hbuf = (float*)(ws + off); off += (size_t)HB_L * HB_D * 4;   //  8 MB
    float*  ubuf = (float*)(ws + off); off += (size_t)HB_M * HB_D * 4;   // 32 MB
    float*  vbuf = (float*)(ws + off); off += (size_t)HB_M * HB_D * 4;   // 32 MB
    __bf16* gbuf = (__bf16*)(ws + off); off += (size_t)HB_M * HB_D * 2;  // 16 MB

    // ---- prep ----
    k_cvt_bf16<<<(HB_M * HB_D) / (256 * 4), 256, 0, stream>>>(x, xb);
    dim3 tgrid(HB_D / 32, HB_D / 32);
    k_transpose_cvt<<<tgrid, 256, 0, stream>>>(Wu, WuT);
    k_transpose_cvt<<<tgrid, 256, 0, stream>>>(Wv, WvT);
    k_transpose_cvt<<<tgrid, 256, 0, stream>>>(Wo, WoT);
    k_decay_h<<<(HB_L * HB_D) / 256, 256, 0, stream>>>(hf, hbuf);

    // ---- u = x@Wu + bu ; v = x@Wv + bv ----
    dim3 ggrid(HB_D / 128, HB_M / 128);
    k_gemm_bias<<<ggrid, 256, 0, stream>>>(xb, WuT, bu, ubuf, HB_M, HB_D, HB_D);
    k_gemm_bias<<<ggrid, 256, 0, stream>>>(xb, WvT, bv, vbuf, HB_M, HB_D, HB_D);

    // ---- v_conv + gate -> bf16 ----
    dim3 cgrid(HB_L / 64, HB_D / 32, HB_B);
    k_conv_gate<<<cgrid, 256, 0, stream>>>(vbuf, hbuf, ubuf, gbuf);

    // ---- y = g@Wo + bo ----
    k_gemm_bias<<<ggrid, 256, 0, stream>>>(gbuf, WoT, bo, out, HB_M, HB_D, HB_D);
}
